// GAT2_51788715655670
// MI455X (gfx1250) — matrix operand order repaired and checked by reading, not compiled, not run on hardware
//
#include <hip/hip_runtime.h>
#include <hip/hip_bf16.h>

typedef __attribute__((ext_vector_type(16))) _Float16 v16h;
typedef __attribute__((ext_vector_type(8)))  _Float16 v8h;
typedef __attribute__((ext_vector_type(2)))  _Float16 v2h;
typedef __attribute__((ext_vector_type(8)))  float    v8f;

#define ATT_SLOPE 0.2f
#define ACT_SLOPE 0.01f
#define BN_EPS    1e-5f
#define NGRAPHS   64

__device__ __forceinline__ float leakyf(float v, float s) { return v >= 0.f ? v : s * v; }

__device__ __forceinline__ void atomic_add_f32(float* p, float v) {
  __hip_atomic_fetch_add(p, v, __ATOMIC_RELAXED, __HIP_MEMORY_SCOPE_AGENT);
}

// ---------------------------------------------------------------------------
// GEMM: C[nrows,256] = A[nrows,256] @ W[256,256], f32 in/out, f16 WMMA compute
// block = 128 threads = 4 waves; each wave: 16 rows x 256 cols (16 WMMA tiles)
// W slab staged in LDS pre-swizzled into B-fragment layout:
//   fragment row r = nt*32 + lane, 16 halfs (K=16*(lane/16)+i, col=nt*16+lane%16),
//   padded to 24 halfs (48B) per row -> each lane loads its fragment with
//   two ds_load_b128 instead of 16 ds_load_u16.
// ---------------------------------------------------------------------------
__global__ __launch_bounds__(128)
void gemm256_wmma(const float* __restrict__ A, const float* __restrict__ W,
                  float* __restrict__ Cout, int nrows)
{
  __shared__ v8h Wl8[16 * 32 * 3];    // 16 tiles * 32 lanes * 24 halfs = 24 KB

  const int tid   = threadIdx.x;
  const int wave  = tid >> 5;
  const int lane  = tid & 31;
  const int lhalf = lane >> 4;        // 0: lanes 0-15, 1: lanes 16-31
  const int l16   = lane & 15;
  const int rowBase = blockIdx.x * 64 + wave * 16;

  v2h* WlP = reinterpret_cast<v2h*>(Wl8);   // packed-pair view (stride 12 v2h/row)

  v8f acc[16] = {};

  for (int kb = 0; kb < 8; ++kb) {
    // ---- stage W[kb*32 .. kb*32+31][0..255] into LDS, fragment-swizzled ----
    // each thread: a 2(K) x 4(N) micro-tile; 1024 groups / 128 threads = 8 iters
    #pragma unroll
    for (int it = 0; it < 8; ++it) {
      const int g  = it * 128 + tid;
      const int kk = (g >> 6) * 2;          // 0,2,..,30 (K within slab)
      const int nn = (g & 63) * 4;          // 0,4,..,252 (column)
      const float* gw = W + (size_t)(kb * 32 + kk) * 256 + nn;
      const float4 w0 = *reinterpret_cast<const float4*>(gw);
      const float4 w1 = *reinterpret_cast<const float4*>(gw + 256);
      const int nt = nn >> 4;
      const int fl = (kk >> 4) * 16 + (nn & 15);   // fragment lane of col nn
      const int i2 = (kk & 15) >> 1;               // packed-pair index in row
      v2h t;
      t[0] = (_Float16)w0.x; t[1] = (_Float16)w1.x;
      WlP[(nt * 32 + fl + 0) * 12 + i2] = t;
      t[0] = (_Float16)w0.y; t[1] = (_Float16)w1.y;
      WlP[(nt * 32 + fl + 1) * 12 + i2] = t;
      t[0] = (_Float16)w0.z; t[1] = (_Float16)w1.z;
      WlP[(nt * 32 + fl + 2) * 12 + i2] = t;
      t[0] = (_Float16)w0.w; t[1] = (_Float16)w1.w;
      WlP[(nt * 32 + fl + 3) * 12 + i2] = t;
    }
    __syncthreads();

    // ---- A fragment (contiguous K = 16*(lane/16)+i, matching B), float4 global loads
    int r = rowBase + l16; if (r >= nrows) r = nrows - 1;
    const float* ap = A + (size_t)r * 256 + kb * 32;
    const float4 a0 = *reinterpret_cast<const float4*>(ap + lhalf * 16);
    const float4 a1 = *reinterpret_cast<const float4*>(ap + lhalf * 16 + 4);
    const float4 a2 = *reinterpret_cast<const float4*>(ap + lhalf * 16 + 8);
    const float4 a3 = *reinterpret_cast<const float4*>(ap + lhalf * 16 + 12);
    v16h afrag;
    afrag[0]  = (_Float16)a0.x; afrag[1]  = (_Float16)a0.y;
    afrag[2]  = (_Float16)a0.z; afrag[3]  = (_Float16)a0.w;
    afrag[4]  = (_Float16)a1.x; afrag[5]  = (_Float16)a1.y;
    afrag[6]  = (_Float16)a1.z; afrag[7]  = (_Float16)a1.w;
    afrag[8]  = (_Float16)a2.x; afrag[9]  = (_Float16)a2.y;
    afrag[10] = (_Float16)a2.z; afrag[11] = (_Float16)a2.w;
    afrag[12] = (_Float16)a3.x; afrag[13] = (_Float16)a3.y;
    afrag[14] = (_Float16)a3.z; afrag[15] = (_Float16)a3.w;

    // ---- 16 column tiles: 2x ds_load_b128 + wmma each ----
    #pragma unroll
    for (int nt = 0; nt < 16; ++nt) {
      const v8h* bp = &Wl8[(nt * 32 + lane) * 3];
      union { struct { v8h lo, hi; } p; v16h v; } u;
      u.p.lo = bp[0];
      u.p.hi = bp[1];
      acc[nt] = __builtin_amdgcn_wmma_f32_16x16x32_f16(
          false, afrag, false, u.v, (short)0, acc[nt], false, false);
    }
    __syncthreads();
  }

  // ---- store C tiles (lane l: N = l%16, M = 8*(l/16)+j) ----
  #pragma unroll
  for (int nt = 0; nt < 16; ++nt) {
    const int col = nt * 16 + l16;
    #pragma unroll
    for (int j = 0; j < 8; ++j) {
      const int row = rowBase + lhalf * 8 + j;
      if (row < nrows) Cout[(size_t)row * 256 + col] = acc[nt][j];
    }
  }
}

// ---------------------------------------------------------------------------
// e_src[i] = h[i] . a_src ; e_dst[i] = h[i] . a_dst   (one node per wave)
// ---------------------------------------------------------------------------
__global__ __launch_bounds__(256)
void node_dots(const float* __restrict__ h, const float* __restrict__ asrc,
               const float* __restrict__ adst, float* __restrict__ es,
               float* __restrict__ ed, int n)
{
  const int wave = threadIdx.x >> 5, lane = threadIdx.x & 31;
  const int node = blockIdx.x * 8 + wave;
  if (node >= n) return;
  const float* hp = h + (size_t)node * 256;
  float s0 = 0.f, s1 = 0.f;
  #pragma unroll
  for (int c = lane; c < 256; c += 32) { float v = hp[c]; s0 += v * asrc[c]; s1 += v * adst[c]; }
  #pragma unroll
  for (int off = 16; off; off >>= 1) {
    s0 += __shfl_xor(s0, off, 32);
    s1 += __shfl_xor(s1, off, 32);
  }
  if (lane == 0) { es[node] = s0; ed[node] = s1; }
}

// order-preserving float<->uint key for atomic max
__device__ __forceinline__ unsigned fkey(float f) {
  unsigned b = __float_as_uint(f);
  return (b & 0x80000000u) ? ~b : (b | 0x80000000u);
}
__device__ __forceinline__ float funkey(unsigned k) {
  return __uint_as_float((k & 0x80000000u) ? (k ^ 0x80000000u) : ~k);
}

// e = leaky(es[src]+ed[dst], 0.2); segment max over dst
__global__ __launch_bounds__(256)
void edge_logits(const long long* __restrict__ ei, const float* __restrict__ es,
                 const float* __restrict__ ed, float* __restrict__ e,
                 unsigned* __restrict__ mkey, int nE, int nN)
{
  const int idx = blockIdx.x * 256 + threadIdx.x;
  const int tot = nE + nN;
  if (idx >= tot) return;
  int s, d;
  if (idx < nE) { s = (int)ei[idx]; d = (int)ei[nE + idx]; }
  else          { s = d = idx - nE; }
  const float v = leakyf(es[s] + ed[d], ATT_SLOPE);
  e[idx] = v;
  atomicMax(&mkey[d], fkey(v));
}

// ex = exp(e - m[dst]); segment sum over dst
__global__ __launch_bounds__(256)
void edge_softmax_num(const long long* __restrict__ ei, float* __restrict__ e,
                      const unsigned* __restrict__ mkey, float* __restrict__ ssum,
                      int nE, int nN)
{
  const int idx = blockIdx.x * 256 + threadIdx.x;
  const int tot = nE + nN;
  if (idx >= tot) return;
  const int d = (idx < nE) ? (int)ei[nE + idx] : (idx - nE);
  const float v = expf(e[idx] - funkey(mkey[d]));
  e[idx] = v;                         // overwrite logits with numerator
  atomic_add_f32(&ssum[d], v);
}

// agg[dst] += (ex/s[dst]) * h[src]  — 64 lanes per edge, 4 channels each
__global__ __launch_bounds__(256)
void edge_aggregate(const long long* __restrict__ ei, const float* __restrict__ ex,
                    const float* __restrict__ ssum, const float* __restrict__ h,
                    float* __restrict__ agg, int nE, int nN)
{
  const int grp = threadIdx.x >> 6;   // edge slot within block (0..3)
  const int t   = threadIdx.x & 63;   // channel group (4 ch each)
  const long long idx = (long long)blockIdx.x * 4 + grp;
  const int tot = nE + nN;
  if (idx >= tot) return;
  int s, d;
  if (idx < nE) { s = (int)ei[idx]; d = (int)ei[nE + idx]; }
  else          { s = d = (int)(idx - nE); }
  const float alpha = ex[idx] / ssum[d];
  const float* hp = h + (size_t)s * 256 + t * 4;
  __builtin_prefetch(hp, 0, 3);
  const float4 hv = *reinterpret_cast<const float4*>(hp);
  float* ap = agg + (size_t)d * 256 + t * 4;
  atomic_add_f32(ap + 0, alpha * hv.x);
  atomic_add_f32(ap + 1, alpha * hv.y);
  atomic_add_f32(ap + 2, alpha * hv.z);
  atomic_add_f32(ap + 3, alpha * hv.w);
}

// h = leaky(h + bias, 0.01) in-place + per-channel sum / sumsq accumulation
__global__ __launch_bounds__(256)
void bias_leaky_stats(float* __restrict__ hio, const float* __restrict__ bias,
                      float* __restrict__ sums, int n, int rowsPerBlock)
{
  const int c = threadIdx.x;
  const float b = bias[c];
  const int r0 = blockIdx.x * rowsPerBlock;
  const int r1 = min(r0 + rowsPerBlock, n);
  float s = 0.f, sq = 0.f;
  for (int r = r0; r < r1; ++r) {
    float v = leakyf(hio[(size_t)r * 256 + c] + b, ACT_SLOPE);
    hio[(size_t)r * 256 + c] = v;
    s += v; sq += v * v;
  }
  if (r0 < n) {
    atomic_add_f32(&sums[c],       s);
    atomic_add_f32(&sums[256 + c], sq);
  }
}

__global__ void bn_coef(const float* __restrict__ sums, const float* __restrict__ gamma,
                        const float* __restrict__ beta, float* __restrict__ coef, int n)
{
  const int c = threadIdx.x;
  const float mean = sums[c] / (float)n;
  const float var  = sums[256 + c] / (float)n - mean * mean;
  const float sc   = gamma[c] * rsqrtf(var + BN_EPS);
  coef[c]       = sc;
  coef[256 + c] = beta[c] - mean * sc;
}

__global__ __launch_bounds__(256)
void bn_apply(const float* __restrict__ hin, const float* __restrict__ coef,
              float* __restrict__ hout, size_t total)
{
  const size_t stride = (size_t)gridDim.x * 256;
  for (size_t i = (size_t)blockIdx.x * 256 + threadIdx.x; i < total; i += stride) {
    const int c = (int)(i & 255);
    hout[i] = hin[i] * coef[c] + coef[256 + c];
  }
}

// v = leaky(agg + bias); pool[batch[r]] += v ; cnt[batch[r]] += 1
__global__ __launch_bounds__(256)
void bias_leaky_pool(const float* __restrict__ agg, const float* __restrict__ bias,
                     const long long* __restrict__ batch, float* __restrict__ pool,
                     float* __restrict__ cnt, int n, int rowsPerBlock)
{
  const int c = threadIdx.x;
  const float b = bias[c];
  const int r0 = blockIdx.x * rowsPerBlock;
  const int r1 = min(r0 + rowsPerBlock, n);
  for (int r = r0; r < r1; ++r) {
    const int g = (int)batch[r];
    const float v = leakyf(agg[(size_t)r * 256 + c] + b, ACT_SLOPE);
    atomic_add_f32(&pool[(size_t)g * 256 + c], v);
    if (c == 0) atomic_add_f32(&cnt[g], 1.0f);
  }
}

// per-graph MLP head: g = pool/cnt; out = leaky(g@lw1+lb1)@lw2 + lb2
__global__ __launch_bounds__(256)
void mlp_head(const float* __restrict__ pool, const float* __restrict__ cnt,
              const float* __restrict__ lw1, const float* __restrict__ lb1,
              const float* __restrict__ lw2, const float* __restrict__ lb2,
              float* __restrict__ out)
{
  __shared__ float gm[256];
  __shared__ float hid[256];
  const int g = blockIdx.x, t = threadIdx.x;
  float c = cnt[g]; c = c > 1.f ? c : 1.f;
  gm[t] = pool[(size_t)g * 256 + t] / c;
  __syncthreads();
  float a = lb1[t];
  for (int k = 0; k < 256; ++k) a += gm[k] * lw1[(size_t)k * 256 + t];
  hid[t] = leakyf(a, ACT_SLOPE);
  __syncthreads();
  if (t < 10) {
    float o = lb2[t];
    for (int k = 0; k < 256; ++k) o += hid[k] * lw2[(size_t)k * 10 + t];
    out[(size_t)g * 10 + t] = o;
  }
}

// ---------------------------------------------------------------------------
extern "C" void kernel_launch(void* const* d_in, const int* in_sizes, int n_in,
                              void* d_out, int out_size, void* d_ws, size_t ws_size,
                              hipStream_t stream)
{
  const float*     x    = (const float*)d_in[0];
  const long long* ei   = (const long long*)d_in[1];   // [2, E] int64
  const long long* batch= (const long long*)d_in[2];   // [N] int64
  const float*     W1   = (const float*)d_in[4];
  const float*     as1  = (const float*)d_in[5];
  const float*     ad1  = (const float*)d_in[6];
  const float*     b1   = (const float*)d_in[7];
  const float*     gamma= (const float*)d_in[8];
  const float*     beta = (const float*)d_in[9];
  const float*     W2   = (const float*)d_in[10];
  const float*     as2  = (const float*)d_in[11];
  const float*     ad2  = (const float*)d_in[12];
  const float*     b2   = (const float*)d_in[13];
  const float*     lw1  = (const float*)d_in[14];
  const float*     lb1  = (const float*)d_in[15];
  const float*     lw2  = (const float*)d_in[16];
  const float*     lb2  = (const float*)d_in[17];
  float* out = (float*)d_out;

  const int Nn = in_sizes[0] / 256;       // 50000
  const int Ee = in_sizes[1] / 2;         // 800000
  const int tot = Ee + Nn;                // edges incl. self loops
  const size_t NH = (size_t)Nn * 256;

  float* ws    = (float*)d_ws;
  float* buf0  = ws;                       // h (layer output of GEMM)
  float* buf1  = buf0 + NH;                // agg / activated
  float* buf2  = buf1 + NH;                // normalized h (conv2 input)
  float* es    = buf2 + NH;
  float* ed    = es + Nn;
  float* mkf   = ed + Nn;                  // uint keys
  float* ssum  = mkf + Nn;
  float* ebuf  = ssum + Nn;                // [E+N] logits -> numerators
  float* stats = ebuf + tot;               // [512]
  float* coef  = stats + 512;              // [512]
  float* pool  = coef + 512;               // [64*256]
  float* cnt   = pool + NGRAPHS * 256;     // [64]
  unsigned* mkey = (unsigned*)mkf;

  const int gemmBlocks = (Nn + 63) / 64;
  const int dotBlocks  = (Nn + 7) / 8;
  const int edgeBlocks = (tot + 255) / 256;
  const int aggBlocks  = (tot + 3) / 4;
  const int statBlocks = 512;
  const int rpb        = (Nn + statBlocks - 1) / statBlocks;

  // ---- layer 1 ----
  hipMemsetAsync(buf1, 0, NH * sizeof(float), stream);
  hipMemsetAsync(mkey, 0, (size_t)Nn * sizeof(unsigned), stream);   // key 0 == -inf
  hipMemsetAsync(ssum, 0, (size_t)Nn * sizeof(float), stream);
  hipMemsetAsync(stats, 0, 512 * sizeof(float), stream);
  hipMemsetAsync(pool, 0, (NGRAPHS * 256 + NGRAPHS) * sizeof(float), stream);

  gemm256_wmma<<<gemmBlocks, 128, 0, stream>>>(x, W1, buf0, Nn);
  node_dots<<<dotBlocks, 256, 0, stream>>>(buf0, as1, ad1, es, ed, Nn);
  edge_logits<<<edgeBlocks, 256, 0, stream>>>(ei, es, ed, ebuf, mkey, Ee, Nn);
  edge_softmax_num<<<edgeBlocks, 256, 0, stream>>>(ei, ebuf, mkey, ssum, Ee, Nn);
  edge_aggregate<<<aggBlocks, 256, 0, stream>>>(ei, ebuf, ssum, buf0, buf1, Ee, Nn);
  bias_leaky_stats<<<statBlocks, 256, 0, stream>>>(buf1, b1, stats, Nn, rpb);
  bn_coef<<<1, 256, 0, stream>>>(stats, gamma, beta, coef, Nn);
  bn_apply<<<4096, 256, 0, stream>>>(buf1, coef, buf2, NH);

  // ---- layer 2 ----
  hipMemsetAsync(buf1, 0, NH * sizeof(float), stream);
  hipMemsetAsync(mkey, 0, (size_t)Nn * sizeof(unsigned), stream);
  hipMemsetAsync(ssum, 0, (size_t)Nn * sizeof(float), stream);

  gemm256_wmma<<<gemmBlocks, 128, 0, stream>>>(buf2, W2, buf0, Nn);
  node_dots<<<dotBlocks, 256, 0, stream>>>(buf0, as2, ad2, es, ed, Nn);
  edge_logits<<<edgeBlocks, 256, 0, stream>>>(ei, es, ed, ebuf, mkey, Ee, Nn);
  edge_softmax_num<<<edgeBlocks, 256, 0, stream>>>(ei, ebuf, mkey, ssum, Ee, Nn);
  edge_aggregate<<<aggBlocks, 256, 0, stream>>>(ei, ebuf, ssum, buf0, buf1, Ee, Nn);
  bias_leaky_pool<<<statBlocks, 256, 0, stream>>>(buf1, b2, batch, pool, cnt, Nn, rpb);

  // ---- head ----
  mlp_head<<<NGRAPHS, 256, 0, stream>>>(pool, cnt, lw1, lb1, lw2, lb2, out);
}
